// VQVAE_37357625541276
// MI455X (gfx1250) — matrix operand order repaired and checked by reading, not compiled, not run on hardware
//
#include <hip/hip_runtime.h>
#include <hip/hip_bf16.h>

// ---------------------------------------------------------------------------
// VQ-VAE quantization for MI455X (gfx1250, wave32, WMMA).
// dist[n,k] = ||c_k||^2 - 2 * x_n . c_k   (||x||^2 dropped: constant per row)
// GEMM done in bf16 with fp32 accumulate via v_wmma_f32_16x16x32_bf16.
// The full next-next code tile is prefetched thread-sliced at WGP scope
// (inline asm: scope 0 pulls into ALL cache levels incl. WGP$) so the
// per-step b128 loads hit at near-L0 latency behind the WMMA in flight.
// ---------------------------------------------------------------------------

typedef __attribute__((ext_vector_type(16))) __bf16        v16bf;
typedef __attribute__((ext_vector_type(8)))  __bf16        v8bf;
typedef __attribute__((ext_vector_type(8)))  float         v8f;
typedef __attribute__((ext_vector_type(4)))  unsigned int  v4u;

#define N_PTS   32768
#define K_CODES 4096
#define D_DIM   512

// d_out layout (floats): Z [N] | q_with_st [N*D] | vq_loss | commitment_loss
#define Z_OFF    0
#define Q_OFF    (N_PTS)
#define LOSS_OFF (N_PTS + N_PTS * D_DIM)

// workspace layout (bytes)
#define WS_CBH_OFF   0                                   // bf16 codebook  4 MB
#define WS_CSQ_OFF   (K_CODES * D_DIM * 2)               // ||c||^2        16 KB
#define WS_ZI_OFF    (WS_CSQ_OFF + K_CODES * 4)          // int Z          128 KB
#define WS_PART_OFF  (WS_ZI_OFF + N_PTS * 4)             // block partials 16 KB
#define N_GATHER_BLK (N_PTS / 8)                         // 4096

// ---------------------------------------------------------------------------
// Kernel 1: per-code ||c||^2 + fp32 -> bf16 codebook conversion.
// One wave per codebook row; lane handles 16 contiguous floats.
// ---------------------------------------------------------------------------
__global__ __launch_bounds__(256) void vq_prep(const float* __restrict__ cb,
                                               __bf16* __restrict__ cbh,
                                               float* __restrict__ csq) {
    const int lane = threadIdx.x & 31;
    const int wave = threadIdx.x >> 5;
    const int k    = blockIdx.x * 8 + wave;                  // 512 blocks * 8
    const float* row = cb + (size_t)k * D_DIM;

    const int o = lane * 16;
    float4 a0 = *(const float4*)(row + o + 0);
    float4 a1 = *(const float4*)(row + o + 4);
    float4 a2 = *(const float4*)(row + o + 8);
    float4 a3 = *(const float4*)(row + o + 12);

    float s = a0.x*a0.x + a0.y*a0.y + a0.z*a0.z + a0.w*a0.w
            + a1.x*a1.x + a1.y*a1.y + a1.z*a1.z + a1.w*a1.w
            + a2.x*a2.x + a2.y*a2.y + a2.z*a2.z + a2.w*a2.w
            + a3.x*a3.x + a3.y*a3.y + a3.z*a3.z + a3.w*a3.w;

    v16bf h;
    h[0]=(__bf16)a0.x;  h[1]=(__bf16)a0.y;  h[2]=(__bf16)a0.z;  h[3]=(__bf16)a0.w;
    h[4]=(__bf16)a1.x;  h[5]=(__bf16)a1.y;  h[6]=(__bf16)a1.z;  h[7]=(__bf16)a1.w;
    h[8]=(__bf16)a2.x;  h[9]=(__bf16)a2.y;  h[10]=(__bf16)a2.z; h[11]=(__bf16)a2.w;
    h[12]=(__bf16)a3.x; h[13]=(__bf16)a3.y; h[14]=(__bf16)a3.z; h[15]=(__bf16)a3.w;
    *(v16bf*)(cbh + (size_t)k * D_DIM + o) = h;

    #pragma unroll
    for (int m = 16; m >= 1; m >>= 1) s += __shfl_xor(s, m, 32);
    if (lane == 0) csq[k] = s;
}

// ---------------------------------------------------------------------------
// Kernel 2: fused bf16-WMMA GEMM + running argmin.
// 512 workgroups x 128 threads (4 waves). WG owns 64 x-rows staged in LDS
// (bf16, XOR-swizzled 16B blocks); the compiler hoists each wave's 16-row
// A-slab into VGPRs. Wave sweeps all 4096 codes, B pipelined across tiles.
// ---------------------------------------------------------------------------
__global__ __launch_bounds__(128) void vq_argmin(const float* __restrict__ x,
                                                 const __bf16* __restrict__ cbh,
                                                 const float* __restrict__ csq,
                                                 float* __restrict__ zf,
                                                 int* __restrict__ zi) {
    __shared__ __align__(16) __bf16 tile[64 * D_DIM];        // 64 KB

    const int t       = threadIdx.x;
    const int rowBase = blockIdx.x * 64;

    // Stage x tile: fp32 -> bf16, 16B blocks, phys_block = b ^ (row & 15).
    #pragma unroll
    for (int i = 0; i < 32; ++i) {
        const int u = t + i * 128;                           // 4096 blocks
        const int r = u >> 6;                                // tile row 0..63
        const int b = u & 63;                                // 8-half block
        const float* src = x + (size_t)(rowBase + r) * D_DIM + b * 8;
        float4 f0 = *(const float4*)(src);
        float4 f1 = *(const float4*)(src + 4);
        v8bf h;
        h[0]=(__bf16)f0.x; h[1]=(__bf16)f0.y; h[2]=(__bf16)f0.z; h[3]=(__bf16)f0.w;
        h[4]=(__bf16)f1.x; h[5]=(__bf16)f1.y; h[6]=(__bf16)f1.z; h[7]=(__bf16)f1.w;
        *(v8bf*)(&tile[r * D_DIM + ((b ^ (r & 15)) << 3)]) = h;
    }
    __syncthreads();

    const int lane    = t & 31;
    const int wave    = t >> 5;
    const int rl      = wave * 16;                 // wave's local row base
    const int rr      = rl + (lane & 15);          // A row this lane feeds
    const int halfsel = lane >> 4;                 // 0: K 0-7/16-23, 1: K 8-15/24-31
    const int swz     = rr & 15;

    float minv[8];
    int   mini[8];
    #pragma unroll
    for (int j = 0; j < 8; ++j) { minv[j] = 3.4e38f; mini[j] = 0; }

    // B fragments, pipelined across code-tile boundaries.
    union BU { v16bf v; v4u u[2]; };
    BU B[2];
    {   // preload (nt=0, kb=0)
        const __bf16* b0 = cbh + (size_t)(lane & 15) * D_DIM + halfsel * 8;
        B[0].u[0] = *(const v4u*)(b0);
        B[0].u[1] = *(const v4u*)(b0 + 16);
    }

    // Per-thread slice of a 16 KB code tile for prefetching: 128 thr * 128 B.
    const char* pfBase = (const char*)cbh + (size_t)t * 128;

    for (int nt = 0; nt < K_CODES / 16; ++nt) {
        const int code = nt * 16 + (lane & 15);
        const float cs = csq[code];
        const __bf16* bp = cbh + (size_t)code * D_DIM + halfsel * 8;

        // Pull the ENTIRE tile nt+2 (16 KB) into ALL cache levels (WGP
        // scope: no scope modifier => SCOPE 0), sliced across the 128
        // threads of the workgroup (128 x 128 B lines).  Trailing
        // iterations prefetch the c_sq/zi ws regions: valid and benign.
        // Prefetch does not touch LOADcnt, so it adds no wait pressure.
        {
            const char* p = pfBase + (size_t)(nt + 2) * (16 * D_DIM * 2);
            asm volatile("global_prefetch_b8 %0, off" :: "v"(p));
        }

        v8f acc = {0.f, 0.f, 0.f, 0.f, 0.f, 0.f, 0.f, 0.f};
        #pragma unroll
        for (int kb = 0; kb < D_DIM / 32; ++kb) {            // 16 WMMAs
            // Issue loads for the NEXT step (next kb, or next tile's kb=0;
            // the final tile's tail load lands in the c_sq region of ws —
            // in-bounds and unused).  16 steps => parity lines up across nt.
            const int nxt = (kb + 1) & 1;
            const __bf16* bn = (kb < (D_DIM / 32 - 1)) ? (bp + (kb + 1) * 32)
                                                       : (bp + 16 * D_DIM);
            B[nxt].u[0] = *(const v4u*)(bn);
            B[nxt].u[1] = *(const v4u*)(bn + 16);

            // A fragment from swizzled LDS (hoisted to VGPRs by compiler).
            const int blk0 = kb * 4 + halfsel;
            union { v16bf v; v4u u[2]; } A;
            A.u[0] = *(const v4u*)&tile[rr * D_DIM + (((blk0    ) ^ swz) << 3)];
            A.u[1] = *(const v4u*)&tile[rr * D_DIM + (((blk0 + 2) ^ swz) << 3)];

            acc = __builtin_amdgcn_wmma_f32_16x16x32_bf16(
                false, A.v, false, B[kb & 1].v, (short)0, acc, false, false);
        }
        #pragma unroll
        for (int j = 0; j < 8; ++j) {
            const float d = cs - 2.0f * acc[j];
            if (d < minv[j]) { minv[j] = d; mini[j] = code; }
        }
    }

    // Reduce across the 16 lanes of each half (C layout: N = lane & 15).
    #pragma unroll
    for (int m = 1; m < 16; m <<= 1) {
        #pragma unroll
        for (int j = 0; j < 8; ++j) {
            const float ov = __shfl_xor(minv[j], m, 32);
            const int   oi = __shfl_xor(mini[j], m, 32);
            if (ov < minv[j] || (ov == minv[j] && oi < mini[j])) {
                minv[j] = ov; mini[j] = oi;
            }
        }
    }
    if ((lane & 15) == 0) {
        const int g = rowBase + rl + halfsel * 8;  // rows j / j+8 per C layout
        #pragma unroll
        for (int j = 0; j < 8; ++j) {
            zf[g + j] = (float)mini[j];
            zi[g + j] = mini[j];
        }
    }
}

// ---------------------------------------------------------------------------
// Kernel 3: gather q = codebook[Z], write x + (q - x), block-partial loss.
// One wave per row; deterministic reduction (no fp atomics).
// ---------------------------------------------------------------------------
__global__ __launch_bounds__(256) void vq_gather(const float* __restrict__ x,
                                                 const float* __restrict__ cb,
                                                 const int* __restrict__ zi,
                                                 float* __restrict__ q,
                                                 float* __restrict__ partials) {
    __shared__ float wsum[8];
    const int lane = threadIdx.x & 31;
    const int wave = threadIdx.x >> 5;
    const int n    = blockIdx.x * 8 + wave;

    const int code   = zi[n];
    const float* cr  = cb + (size_t)code * D_DIM;
    const float* xr  = x  + (size_t)n * D_DIM;
    float*       qr  = q  + (size_t)n * D_DIM;

    float s = 0.f;
    #pragma unroll
    for (int i = 0; i < 4; ++i) {
        const int o = lane * 16 + i * 4;
        float4 c4 = *(const float4*)(cr + o);
        float4 x4 = *(const float4*)(xr + o);
        float4 d4 = make_float4(c4.x - x4.x, c4.y - x4.y, c4.z - x4.z, c4.w - x4.w);
        s += d4.x*d4.x + d4.y*d4.y + d4.z*d4.z + d4.w*d4.w;
        // straight-through value, same rounding as reference: x + (q - x)
        float4 o4 = make_float4(x4.x + d4.x, x4.y + d4.y, x4.z + d4.z, x4.w + d4.w);
        *(float4*)(qr + o) = o4;
    }
    #pragma unroll
    for (int m = 16; m >= 1; m >>= 1) s += __shfl_xor(s, m, 32);
    if (lane == 0) wsum[wave] = s;
    __syncthreads();
    if (threadIdx.x == 0) {
        float tot = 0.f;
        #pragma unroll
        for (int i = 0; i < 8; ++i) tot += wsum[i];          // fixed order
        partials[blockIdx.x] = tot;
    }
}

// ---------------------------------------------------------------------------
// Kernel 4: deterministic final reduction -> vq_loss == commitment_loss.
// ---------------------------------------------------------------------------
__global__ __launch_bounds__(256) void vq_final(const float* __restrict__ partials,
                                                float* __restrict__ out) {
    __shared__ float sh[256];
    const int t = threadIdx.x;
    float s = 0.f;
    #pragma unroll
    for (int i = 0; i < N_GATHER_BLK / 256; ++i) s += partials[t + i * 256];
    sh[t] = s;
    __syncthreads();
    for (int st = 128; st > 0; st >>= 1) {
        if (t < st) sh[t] += sh[t + st];
        __syncthreads();
    }
    if (t == 0) {
        const float m = sh[0] * (1.0f / ((float)N_PTS * (float)D_DIM));
        out[0] = m;   // vq_loss
        out[1] = m;   // commitment_loss (numerically identical)
    }
}

// ---------------------------------------------------------------------------
extern "C" void kernel_launch(void* const* d_in, const int* in_sizes, int n_in,
                              void* d_out, int out_size, void* d_ws, size_t ws_size,
                              hipStream_t stream) {
    (void)in_sizes; (void)n_in; (void)out_size; (void)ws_size;

    const float* x  = (const float*)d_in[0];
    const float* cb = (const float*)d_in[1];
    float* out      = (float*)d_out;

    char* ws = (char*)d_ws;
    __bf16* cbh    = (__bf16*)(ws + WS_CBH_OFF);
    float*  csq    = (float*) (ws + WS_CSQ_OFF);
    int*    zi     = (int*)   (ws + WS_ZI_OFF);
    float*  parts  = (float*) (ws + WS_PART_OFF);

    vq_prep  <<<K_CODES / 8,   256, 0, stream>>>(cb, cbh, csq);
    vq_argmin<<<N_PTS / 64,    128, 0, stream>>>(x, cbh, csq, out + Z_OFF, zi);
    vq_gather<<<N_GATHER_BLK,  256, 0, stream>>>(x, cb, zi, out + Q_OFF, parts);
    vq_final <<<1,             256, 0, stream>>>(parts, out + LOSS_OFF);
}
